// MultiEmbodimentActionEncoder_60473139528226
// MI455X (gfx1250) — matrix operand order repaired and checked by reading, not compiled, not run on hardware
//
#include <hip/hip_runtime.h>

typedef __attribute__((ext_vector_type(16))) _Float16 v16h;
typedef __attribute__((ext_vector_type(8)))  _Float16 v8h;
typedef __attribute__((ext_vector_type(8)))  float    v8f;
typedef __attribute__((ext_vector_type(2)))  float    v2f;

#define BB   256
#define TT   64
#define AAC  32
#define HH   512
#define EE   32
#define XSTR 520   // padded LDS row stride (halfs): 1040B rows -> 16B aligned, 4-bank skew

// ---- packed f16 weight-fragment buffers in workspace ----
// Fragment tile = 32x16 (KxN) B-operand, stored lane-major:
//   halfs[((catTile)*KS + ks)*32*16 + lane*16 + j]
//   element = W[ks*32 + 16*(lane>>4) + j][ntile*16 + (lane&15)]
#define FW1_HALFS ((size_t)EE * 32 * 1  * 512)   // 524,288  (1 MB)
#define FW2_HALFS ((size_t)EE * 32 * 16 * 512)   // 8,388,608 (16 MB)
#define FW3_HALFS ((size_t)EE * 32 * 16 * 512)   // 8,388,608 (16 MB)
#define WS_NEED   ((FW1_HALFS + FW2_HALFS + FW3_HALFS) * sizeof(_Float16))

// ---------------- weight packing pre-pass: fp32 -> f16 fragment order ----------------
// 1056 tiles per cat (32 W1 + 512 W2top + 512 W3), 32 cats = 33792 tiles, 1 wave/tile.
__global__ __launch_bounds__(256) void pack_weights(
    const float* __restrict__ W1,   // (E,32,512)
    const float* __restrict__ W2,   // (E,1024,512)  (top 512 rows packed)
    const float* __restrict__ W3,   // (E,512,512)
    _Float16* __restrict__ fW1, _Float16* __restrict__ fW2, _Float16* __restrict__ fW3)
{
    const int lane  = threadIdx.x & 31;
    const int wave  = threadIdx.x >> 5;
    const int tile  = blockIdx.x * 8 + wave;        // 0 .. 33791
    const int cat   = tile / 1056;
    const int r     = tile % 1056;
    const int l16   = lane & 15;
    const int lhalf = lane >> 4;

    const float* src;
    _Float16*    dst;
    int nt, ks, KS;
    if (r < 32) {                 // W1: K=32, 1 k-step
        nt = r; ks = 0; KS = 1;
        src = W1 + (size_t)cat * AAC * HH;
        dst = fW1;
    } else if (r < 544) {         // W2 top half: K=512, 16 k-steps
        int q = r - 32; nt = q >> 4; ks = q & 15; KS = 16;
        src = W2 + (size_t)cat * (2 * HH) * HH;
        dst = fW2;
    } else {                      // W3: K=512, 16 k-steps
        int q = r - 544; nt = q >> 4; ks = q & 15; KS = 16;
        src = W3 + (size_t)cat * HH * HH;
        dst = fW3;
    }

    const float* p = src + (size_t)(ks * 32 + lhalf * 16) * HH + nt * 16 + l16;
    v16h f;
    #pragma unroll
    for (int j = 0; j < 16; ++j) f[j] = (_Float16)p[(size_t)j * HH];

    const size_t off = (((size_t)(cat * 32 + nt) * KS + ks) * 32 + lane) * 16;
    *(v16h*)(dst + off) = f;
}

// ---------------- main fused encoder kernel ----------------
template <bool FAST>
__global__ __launch_bounds__(256) void enc_kernel(
    const float* __restrict__ actions,   // (B,T,A)
    const int*   __restrict__ timesteps, // (B,)
    const int*   __restrict__ cat_ids,   // (B,)
    const float* __restrict__ W1,        // (E,A,H)
    const float* __restrict__ b1,        // (E,H)
    const float* __restrict__ W2,        // (E,2H,H)
    const float* __restrict__ b2,        // (E,H)
    const float* __restrict__ W3,        // (E,H,H)
    const float* __restrict__ b3,        // (E,H)
    const _Float16* __restrict__ fW1,    // packed fragments (FAST only)
    const _Float16* __restrict__ fW2,
    const _Float16* __restrict__ fW3,
    float* __restrict__ out)             // (B,T,H)
{
    __shared__ __align__(16) _Float16 xbuf[TT * XSTR]; // a_emb (f16), then reused for h
    __shared__ float tv[HH];      // sinusoidal tau vector (per-sample, constant over T)
    __shared__ float hinit[HH];   // b2 + tau @ W2[:,512:,:]  (folded accumulator init)

    const int b    = blockIdx.x;
    const int tid  = threadIdx.x;
    const int lane = tid & 31;
    const int wave = tid >> 5;        // 0..7
    const int l16  = lane & 15;
    const int lhalf = lane >> 4;      // 0 or 1

    const int   cat   = cat_ids[b];
    const float tstep = (float)timesteps[b];

    // ---- tau vector: tv[i] = sin(t*f_i) for i<256, cos(t*f_{i-256}) otherwise ----
    {
        const float cexp = -logf(10000.0f) / 256.0f;
        for (int i = tid; i < HH; i += 256) {
            int j = (i < 256) ? i : (i - 256);
            float ph = tstep * __expf(cexp * (float)j);
            tv[i] = (i < 256) ? __sinf(ph) : __cosf(ph);
        }
    }
    __syncthreads();

    // Slow path: gather B-fragment from row-major fp32 weights, convert to f16.
    auto make_bfrag = [&](const float* Wbase, int kbase, int ncol) -> v16h {
        const float* p = Wbase + (size_t)(kbase + lhalf * 16) * HH + ncol;
        v16h bf;
        #pragma unroll
        for (int j = 0; j < 16; ++j) bf[j] = (_Float16)p[(size_t)j * HH];
        return bf;
    };
    // Fast path: one coalesced 32B load per lane from pre-packed fragments.
    auto load_bfrag = [&](const _Float16* fW, int KS, int ntAbs, int ks) -> v16h {
        return *(const v16h*)(fW + (((size_t)(cat * 32 + ntAbs) * KS + ks) * 32 + lane) * 16);
    };

    // ---- Layer 1: a_emb = actions @ W1[cat] + b1[cat]  -> f16 in xbuf ----
    {
        const float* W1c = W1 + (size_t)cat * AAC * HH;
        const float* b1c = b1 + (size_t)cat * HH;
        const int lo = lhalf * 16;
        v16h afr[4];
        #pragma unroll
        for (int mt = 0; mt < 4; ++mt) {
            const float* ar = actions + ((size_t)b * TT + mt * 16 + l16) * AAC;
            v16h a;
            #pragma unroll
            for (int j = 0; j < 8; ++j) { a[j] = (_Float16)ar[lo + j]; a[8 + j] = (_Float16)ar[lo + 8 + j]; }
            afr[mt] = a;
        }
        #pragma unroll
        for (int nt = 0; nt < 4; ++nt) {
            const int ntAbs = wave * 4 + nt;
            const int ncol  = ntAbs * 16 + l16;
            v16h bfr;
            if constexpr (FAST) bfr = load_bfrag(fW1, 1, ntAbs, 0);
            else                bfr = make_bfrag(W1c, 0, ncol);
            const float bias = b1c[ncol];
            #pragma unroll
            for (int mt = 0; mt < 4; ++mt) {
                v8f c;
                #pragma unroll
                for (int r = 0; r < 8; ++r) c[r] = bias;
                c = __builtin_amdgcn_wmma_f32_16x16x32_f16(false, afr[mt], false, bfr,
                                                           (short)0, c, false, false);
                const int mrow = mt * 16 + lhalf * 8;
                #pragma unroll
                for (int r = 0; r < 8; ++r)
                    xbuf[(mrow + r) * XSTR + ncol] = (_Float16)c[r];
            }
        }
    }

    // ---- hinit = b2[cat] + tau @ W2[cat, 512:1024, :]  (mat-vec, thread-parallel) ----
    {
        const float* W2bot = W2 + (size_t)cat * (2 * HH) * HH + (size_t)HH * HH;
        const int n0 = tid * 2;
        float a0 = 0.f, a1 = 0.f;
        for (int k = 0; k < HH; ++k) {
            const float tvk = tv[k];
            const v2f w = *(const v2f*)(W2bot + (size_t)k * HH + n0);
            a0 += tvk * w[0];
            a1 += tvk * w[1];
        }
        hinit[n0]     = a0 + b2[(size_t)cat * HH + n0];
        hinit[n0 + 1] = a1 + b2[(size_t)cat * HH + n0 + 1];
    }
    __syncthreads();

    // A-fragment from LDS xbuf: two aligned 16B (v8h) chunks per fragment.
    auto make_afrag_lds = [&](int row, int k0) -> v16h {
        const v8h lo8 = *(const v8h*)&xbuf[row * XSTR + k0 + lhalf * 16];
        const v8h hi8 = *(const v8h*)&xbuf[row * XSTR + k0 + lhalf * 16 + 8];
        v16h a;
        #pragma unroll
        for (int j = 0; j < 8; ++j) { a[j] = lo8[j]; a[8 + j] = hi8[j]; }
        return a;
    };

    // ---- Layer 2: h = swish(a_emb @ W2_top + hinit)  -> f16 back into xbuf ----
    {
        const float* W2t = W2 + (size_t)cat * (2 * HH) * HH;
        v8f acc[4][4];
        #pragma unroll
        for (int nt = 0; nt < 4; ++nt) {
            const float hi = hinit[(wave * 4 + nt) * 16 + l16];
            #pragma unroll
            for (int mt = 0; mt < 4; ++mt)
                #pragma unroll
                for (int r = 0; r < 8; ++r) acc[mt][nt][r] = hi;
        }
        for (int k0 = 0; k0 < HH; k0 += 32) {
            v16h afr[4];
            #pragma unroll
            for (int mt = 0; mt < 4; ++mt) afr[mt] = make_afrag_lds(mt * 16 + l16, k0);
            #pragma unroll
            for (int nt = 0; nt < 4; ++nt) {
                const int ntAbs = wave * 4 + nt;
                v16h bfr;
                if constexpr (FAST) bfr = load_bfrag(fW2, 16, ntAbs, k0 >> 5);
                else                bfr = make_bfrag(W2t, k0, ntAbs * 16 + l16);
                #pragma unroll
                for (int mt = 0; mt < 4; ++mt)
                    acc[mt][nt] = __builtin_amdgcn_wmma_f32_16x16x32_f16(false, afr[mt], false, bfr,
                                                                         (short)0, acc[mt][nt], false, false);
            }
        }
        __syncthreads();   // all xbuf reads complete before overwrite
        #pragma unroll
        for (int nt = 0; nt < 4; ++nt) {
            const int ncol = (wave * 4 + nt) * 16 + l16;
            #pragma unroll
            for (int mt = 0; mt < 4; ++mt) {
                const int mrow = mt * 16 + lhalf * 8;
                #pragma unroll
                for (int r = 0; r < 8; ++r) {
                    const float v = acc[mt][nt][r];
                    const float s = v / (1.0f + __expf(-v));   // swish
                    xbuf[(mrow + r) * XSTR + ncol] = (_Float16)s;
                }
            }
        }
    }
    __syncthreads();

    // ---- Layer 3: out = h @ W3[cat] + b3[cat]  (fp32 to global) ----
    {
        const float* W3c = W3 + (size_t)cat * HH * HH;
        const float* b3c = b3 + (size_t)cat * HH;
        v8f acc[4][4];
        #pragma unroll
        for (int nt = 0; nt < 4; ++nt) {
            const float bias = b3c[(wave * 4 + nt) * 16 + l16];
            #pragma unroll
            for (int mt = 0; mt < 4; ++mt)
                #pragma unroll
                for (int r = 0; r < 8; ++r) acc[mt][nt][r] = bias;
        }
        for (int k0 = 0; k0 < HH; k0 += 32) {
            v16h afr[4];
            #pragma unroll
            for (int mt = 0; mt < 4; ++mt) afr[mt] = make_afrag_lds(mt * 16 + l16, k0);
            #pragma unroll
            for (int nt = 0; nt < 4; ++nt) {
                const int ntAbs = wave * 4 + nt;
                v16h bfr;
                if constexpr (FAST) bfr = load_bfrag(fW3, 16, ntAbs, k0 >> 5);
                else                bfr = make_bfrag(W3c, k0, ntAbs * 16 + l16);
                #pragma unroll
                for (int mt = 0; mt < 4; ++mt)
                    acc[mt][nt] = __builtin_amdgcn_wmma_f32_16x16x32_f16(false, afr[mt], false, bfr,
                                                                         (short)0, acc[mt][nt], false, false);
            }
        }
        #pragma unroll
        for (int nt = 0; nt < 4; ++nt) {
            const int ncol = (wave * 4 + nt) * 16 + l16;
            #pragma unroll
            for (int mt = 0; mt < 4; ++mt) {
                const int mrow = mt * 16 + lhalf * 8;
                #pragma unroll
                for (int r = 0; r < 8; ++r)
                    out[((size_t)b * TT + mrow + r) * HH + ncol] = acc[mt][nt][r];
            }
        }
    }
}

extern "C" void kernel_launch(void* const* d_in, const int* in_sizes, int n_in,
                              void* d_out, int out_size, void* d_ws, size_t ws_size,
                              hipStream_t stream) {
    (void)in_sizes; (void)n_in; (void)out_size;
    const float* actions   = (const float*)d_in[0];
    const int*   timesteps = (const int*)d_in[1];
    const int*   cat_ids   = (const int*)d_in[2];
    const float* W1 = (const float*)d_in[3];
    const float* b1 = (const float*)d_in[4];
    const float* W2 = (const float*)d_in[5];
    const float* b2 = (const float*)d_in[6];
    const float* W3 = (const float*)d_in[7];
    const float* b3 = (const float*)d_in[8];
    float* out = (float*)d_out;

    if (ws_size >= WS_NEED && d_ws != nullptr) {
        _Float16* fW1 = (_Float16*)d_ws;
        _Float16* fW2 = fW1 + FW1_HALFS;
        _Float16* fW3 = fW2 + FW2_HALFS;
        // 33792 fragment tiles, 8 waves/block -> 4224 blocks
        pack_weights<<<dim3(4224), dim3(256), 0, stream>>>(W1, W2, W3, fW1, fW2, fW3);
        enc_kernel<true><<<dim3(BB), dim3(256), 0, stream>>>(actions, timesteps, cat_ids,
                                                             W1, b1, W2, b2, W3, b3,
                                                             fW1, fW2, fW3, out);
    } else {
        enc_kernel<false><<<dim3(BB), dim3(256), 0, stream>>>(actions, timesteps, cat_ids,
                                                              W1, b1, W2, b2, W3, b3,
                                                              nullptr, nullptr, nullptr, out);
    }
}
